// EnhancedEMA_20521353740560
// MI455X (gfx1250) — matrix operand order repaired and checked by reading, not compiled, not run on hardware
//
#include <hip/hip_runtime.h>
#include <hip/hip_bf16.h>

typedef _Float16 v16h __attribute__((ext_vector_type(16)));
typedef float    v8f  __attribute__((ext_vector_type(8)));

#define Bn 32
#define Tn 2048
#define Cn 512
#define Kn 512  // T/4

// workspace layout in floats
#define WS_PARTIAL 0        // 1024 : per-(b,chunk) partial sums for mean
#define WS_M       1024     // 32   : per-batch mean
#define WS_RS      2048     // 512  : row-sums of w1
#define WS_H1      4096     // 16384: h1 = relu(m[b]*rs[j] + b1[j])  (32 x 512)
#define WS_LOGITS  20480    // 65536: logits, overwritten in-place by alpha (32 x 2048)

// ---------------- K1: partial sums of x for per-batch mean ----------------
// grid (32, 32) blocks of 256; block (b,i) sums a 32768-element slice of batch b.
__global__ void k_mean_partial(const float* __restrict__ x, float* __restrict__ ws) {
    __shared__ float sdata[256];
    const int b = blockIdx.x, i = blockIdx.y, tid = threadIdx.x;
    const float4* xb = (const float4*)(x + (size_t)b * (Tn * Cn) + (size_t)i * 32768);
    float s = 0.f;
#pragma unroll
    for (int k = 0; k < 32; ++k) {                 // 32 * 256 float4 = 32768 floats
        float4 v = xb[k * 256 + tid];
        s += v.x + v.y + v.z + v.w;
    }
    sdata[tid] = s;
    __syncthreads();
    for (int r = 128; r > 0; r >>= 1) {
        if (tid < r) sdata[tid] += sdata[tid + r];
        __syncthreads();
    }
    if (tid == 0) ws[WS_PARTIAL + b * 32 + i] = sdata[0];
}

// ---------------- K2: finalize means (1 block, 32 threads) ----------------
__global__ void k_mean_final(float* __restrict__ ws) {
    const int b = threadIdx.x;
    float s = 0.f;
#pragma unroll
    for (int i = 0; i < 32; ++i) s += ws[WS_PARTIAL + b * 32 + i];
    ws[WS_M + b] = s * (1.0f / (float)(Tn * Cn));
}

// ---------------- K3: row-sums of w1 (h is constant per row!) ----------------
// grid 512 blocks of 256; block j sums w1[j, :] (2048 floats).
__global__ void k_w1_rowsum(const float* __restrict__ w1, float* __restrict__ ws) {
    __shared__ float sdata[256];
    const int j = blockIdx.x, tid = threadIdx.x;
    const float4* row = (const float4*)(w1 + (size_t)j * Tn);
    float4 a = row[tid];
    float4 b = row[tid + 256];
    sdata[tid] = a.x + a.y + a.z + a.w + b.x + b.y + b.z + b.w;
    __syncthreads();
    for (int r = 128; r > 0; r >>= 1) {
        if (tid < r) sdata[tid] += sdata[tid + r];
        __syncthreads();
    }
    if (tid == 0) ws[WS_RS + j] = sdata[0];
}

// ---------------- K4: h1[b,j] = relu(m[b]*rs[j] + b1[j]) ----------------
__global__ void k_h1(const float* __restrict__ b1, float* __restrict__ ws) {
    const int idx = blockIdx.x * blockDim.x + threadIdx.x;   // 16384
    const int b = idx >> 9, j = idx & 511;
    float v = ws[WS_M + b] * ws[WS_RS + j] + b1[j];
    ws[WS_H1 + idx] = v > 0.f ? v : 0.f;
}

// ---------------- K5: logits = h1 @ w2^T + b2 via WMMA ----------------
// M=32 (batch), N=2048 (t), K=512. 2x128 = 256 tiles of 16x16; one wave per tile.
// grid 64 blocks of 128 (4 waves/block).
__global__ void k_logits_wmma(const float* __restrict__ w2, const float* __restrict__ b2,
                              float* __restrict__ ws) {
    const int wave = (blockIdx.x * blockDim.x + threadIdx.x) >> 5;  // 0..255
    const int lane = threadIdx.x & 31;
    const int half = lane >> 4;          // 0: lanes 0-15, 1: lanes 16-31
    const int l16  = lane & 15;
    const int tm = wave & 1;             // 2 M-tiles
    const int tn = wave >> 1;            // 128 N-tiles
    const int rowbase = tm * 16;         // batch row
    const int colbase = tn * 16;         // t column

    const float* __restrict__ hrow = ws + WS_H1 + (size_t)(rowbase + l16) * Kn;
    const float* __restrict__ wrow = w2 + (size_t)(colbase + l16) * Kn;   // w2[t, j]

    v8f acc = {};
    for (int k0 = 0; k0 < Kn; k0 += 32) {
        v16h a, bm;
        // A (16x32 f16): lane half 0 holds K 0..7 & 16..23, half 1 holds 8..15 & 24..31
#pragma unroll
        for (int e = 0; e < 8; ++e) {
            a[e]     = (_Float16)hrow[k0 + half * 8 + e];
            a[e + 8] = (_Float16)hrow[k0 + 16 + half * 8 + e];
        }
        // B (32x16 f16): lane = column; half 0 holds K 0..7 & 16..23, half 1 holds 8..15 & 24..31
#pragma unroll
        for (int e = 0; e < 8; ++e) {
            bm[e]     = (_Float16)wrow[k0 + half * 8 + e];
            bm[e + 8] = (_Float16)wrow[k0 + 16 + half * 8 + e];
        }

        acc = __builtin_amdgcn_wmma_f32_16x16x32_f16(
            /*neg_a=*/false, a, /*neg_b=*/false, bm,
            /*c_mod=*/(short)0, acc, /*reuse_a=*/false, /*reuse_b=*/false);
    }

    // C/D layout: VGPR v -> row v + 8*half, col = l16
    const int col = colbase + l16;
    const float bias = b2[col];
#pragma unroll
    for (int v = 0; v < 8; ++v) {
        const int row = rowbase + v + 8 * half;
        ws[WS_LOGITS + (size_t)row * Tn + col] = acc[v] + bias;
    }
}

// ---------------- K6: softmax over t, times position_alpha (in place) ----------------
// grid 32 blocks (one per batch) of 256 threads.
__global__ void k_softmax_alpha(const float* __restrict__ pa, float* __restrict__ ws) {
    __shared__ float vals[Tn];     // 8 KB
    __shared__ float red[256];
    const int b = blockIdx.x, tid = threadIdx.x;
    float* __restrict__ logits = ws + WS_LOGITS + (size_t)b * Tn;

    float mx = -3.4e38f;
#pragma unroll
    for (int r = 0; r < 8; ++r) {
        float v = logits[tid + r * 256];
        vals[tid + r * 256] = v;
        mx = fmaxf(mx, v);
    }
    red[tid] = mx;
    __syncthreads();
    for (int r = 128; r > 0; r >>= 1) {
        if (tid < r) red[tid] = fmaxf(red[tid], red[tid + r]);
        __syncthreads();
    }
    mx = red[0];
    __syncthreads();

    float sum = 0.f;
#pragma unroll
    for (int r = 0; r < 8; ++r) {
        float e = __expf(vals[tid + r * 256] - mx);
        vals[tid + r * 256] = e;
        sum += e;
    }
    red[tid] = sum;
    __syncthreads();
    for (int r = 128; r > 0; r >>= 1) {
        if (tid < r) red[tid] += red[tid + r];
        __syncthreads();
    }
    const float inv = 1.0f / red[0];
    __syncthreads();

#pragma unroll
    for (int r = 0; r < 8; ++r) {
        const int t = tid + r * 256;
        logits[t] = pa[t] * vals[t] * inv;   // alpha[b,t], in place
    }
}

// ---------------- K7: adaptive EMA scan over T ----------------
// grid 64 blocks of 256: 2 blocks per batch, one lane per channel c.
// Coalesced 128B/wave loads of x; alpha row staged in LDS; prefetch ahead.
__global__ void k_ema(const float* __restrict__ x, const float* __restrict__ ws,
                      float* __restrict__ out) {
    __shared__ float al[Tn];       // 8 KB
    const int b = blockIdx.x >> 1;
    const int c = ((blockIdx.x & 1) << 8) + threadIdx.x;
    const float* __restrict__ arow = ws + WS_LOGITS + (size_t)b * Tn;
#pragma unroll
    for (int r = 0; r < 8; ++r) al[threadIdx.x + r * 256] = arow[threadIdx.x + r * 256];
    __syncthreads();

    const size_t base = (size_t)b * Tn * Cn + c;
    const float* __restrict__ xp = x + base;
    float* __restrict__ op = out + base;

    float s = xp[0];
    op[0] = s;
#pragma unroll 4
    for (int t = 1; t < Tn; ++t) {
        if (t + 16 < Tn) __builtin_prefetch(&xp[(size_t)(t + 16) * Cn], 0, 1);
        const float a  = al[t];
        const float xv = xp[(size_t)t * Cn];
        s = fmaf(a, xv - s, s);            // a*x + (1-a)*s
        op[(size_t)t * Cn] = s;
    }
}

extern "C" void kernel_launch(void* const* d_in, const int* in_sizes, int n_in,
                              void* d_out, int out_size, void* d_ws, size_t ws_size,
                              hipStream_t stream) {
    const float* x  = (const float*)d_in[0];
    const float* pa = (const float*)d_in[1];
    const float* w1 = (const float*)d_in[2];
    const float* b1 = (const float*)d_in[3];
    const float* w2 = (const float*)d_in[4];
    const float* b2 = (const float*)d_in[5];
    float* out = (float*)d_out;
    float* ws  = (float*)d_ws;

    k_mean_partial<<<dim3(32, 32), 256, 0, stream>>>(x, ws);
    k_mean_final  <<<1, 32, 0, stream>>>(ws);
    k_w1_rowsum   <<<512, 256, 0, stream>>>(w1, ws);
    k_h1          <<<64, 256, 0, stream>>>(b1, ws);
    k_logits_wmma <<<64, 128, 0, stream>>>(w2, b2, ws);
    k_softmax_alpha<<<32, 256, 0, stream>>>(pa, ws);
    k_ema         <<<64, 256, 0, stream>>>(x, ws, out);
}
